// MultiHeadAttention_670014898269
// MI455X (gfx1250) — matrix operand order repaired and checked by reading, not compiled, not run on hardware
//
#include <hip/hip_runtime.h>
#include <hip/hip_bf16.h>

#define D_MODELC 1024
#define NUM_HEADSC 16
#define D_KC 64
#define BATCHC 4
#define SEQC 2048
#define M_TOTAL (BATCHC * SEQC) // 8192

typedef _Float16 v8h  __attribute__((ext_vector_type(8)));
typedef _Float16 v16h __attribute__((ext_vector_type(16)));
typedef float    v8f  __attribute__((ext_vector_type(8)));
typedef int      v4i  __attribute__((ext_vector_type(4)));

union F16x16 { v16h v; v8h h[2]; };

__device__ __forceinline__ v16h ld16(const _Float16* p0, const _Float16* p1) {
    F16x16 u;
    u.h[0] = *(const v8h*)p0;
    u.h[1] = *(const v8h*)p1;
    return u.v;
}

__device__ __forceinline__ v8f wmma_f16f32(v16h a, v16h b, v8f c) {
    // v_wmma_f32_16x16x32_f16: D = A(16x32 f16) * B(32x16 f16) + C(16x16 f32)
    return __builtin_amdgcn_wmma_f32_16x16x32_f16(
        /*neg_a=*/false, a, /*neg_b=*/false, b,
        /*c_mod=*/(short)0, c, /*reuse_a=*/false, /*reuse_b=*/false);
}

__device__ __forceinline__ v8f zero_v8f() {
    v8f z = {0.f, 0.f, 0.f, 0.f, 0.f, 0.f, 0.f, 0.f};
    return z;
}

// ---------------------------------------------------------------------------
// CDNA5 async global->LDS staging (ASYNCcnt-tracked DMA, bypasses VGPRs).
// The builtin takes (global v4i*, shared v4i*, imm offset, imm cpol).
// Guarded: falls back to the proven load->ds_store path if the builtins are
// not declared by this toolchain.
// ---------------------------------------------------------------------------
#if defined(__has_builtin)
#if __has_builtin(__builtin_amdgcn_global_load_async_to_lds_b128) && \
    __has_builtin(__builtin_amdgcn_s_wait_asynccnt)
#define HAVE_ASYNC_LDS 1
#endif
#endif
#ifndef HAVE_ASYNC_LDS
#define HAVE_ASYNC_LDS 0
#endif

typedef __attribute__((address_space(1))) v4i* gptr_v4i;
typedef __attribute__((address_space(3))) v4i* lptr_v4i;

__device__ __forceinline__ void cp16_async(const _Float16* g, _Float16* l) {
#if HAVE_ASYNC_LDS
    __builtin_amdgcn_global_load_async_to_lds_b128(
        (gptr_v4i)g, (lptr_v4i)l, /*offset=*/0, /*cpol=*/0);
#else
    *(v8h*)l = *(const v8h*)g;
#endif
}

__device__ __forceinline__ void async_wait0() {
#if HAVE_ASYNC_LDS
    __builtin_amdgcn_s_wait_asynccnt(0);
#endif
}

// ---------------------------------------------------------------------------
// fp32 -> f16 conversion (activations and weights), 4 elems/thread
// ---------------------------------------------------------------------------
__global__ void cvt_f32_f16(const float* __restrict__ src,
                            _Float16* __restrict__ dst, int n) {
    int i = (blockIdx.x * blockDim.x + threadIdx.x) * 4;
    if (i + 3 >= n) return;
    float4 f = *(const float4*)(src + i);
    dst[i + 0] = (_Float16)f.x;
    dst[i + 1] = (_Float16)f.y;
    dst[i + 2] = (_Float16)f.z;
    dst[i + 3] = (_Float16)f.w;
}

// ---------------------------------------------------------------------------
// Y = X @ W^T + b   (X:[M,1024] f16, W:[1024,1024] f16 row-major)
// mode 0: write Qh  [B,H,S,DK] f16
// mode 1: write Kh  [B,H,S,DK] f16
// mode 2: write Vt  [B,H,DK,S] f16 (transposed per head for PV B-fragments)
// mode 3: write out [M,1024]  f32 (+bias)  -- final projection
// Block tile 128x128x32, 8 waves: 2(M) x 4(N); wave tile 64x32 = 4x2 WMMA.
// Double-buffered LDS; next K-slice staged by async DMA while WMMAs run.
// ---------------------------------------------------------------------------
#define BM 128
#define BN 128
#define BK 32
#define LDT 40 // padded LDS row stride (f16): 80B, 16B aligned, conflict-skewed
#define GEMM_BUF_F16 (BM * LDT + BN * LDT)        // one A+W buffer
#define GEMM_SMEM_F16 (2 * GEMM_BUF_F16)          // double buffered

__device__ __forceinline__ void stage_gemm_tile(
    const _Float16* __restrict__ X, const _Float16* __restrict__ W,
    _Float16* As, _Float16* Ws, int m0, int n0, int k0, int tid) {
#pragma unroll
    for (int c = 0; c < 2; ++c) {
        int chunk = tid + c * 256;
        int row = chunk >> 2;
        int cc  = (chunk & 3) * 8;
        cp16_async(X + (size_t)(m0 + row) * D_MODELC + k0 + cc, As + row * LDT + cc);
        cp16_async(W + (size_t)(n0 + row) * D_MODELC + k0 + cc, Ws + row * LDT + cc);
    }
}

__global__ __launch_bounds__(256) void gemm_xwT(
    const _Float16* __restrict__ X,
    const _Float16* __restrict__ W,
    const float* __restrict__ bias,
    _Float16* __restrict__ out_h,
    float* __restrict__ out_f,
    int mode)
{
    extern __shared__ _Float16 gsm[];

    const int tid  = threadIdx.x;
    const int lane = tid & 31;
    const int wv   = tid >> 5;
    const int wm   = wv & 1;   // 2 waves along M
    const int wn   = wv >> 1;  // 4 waves along N
    const int hf   = lane >> 4;
    const int cl   = lane & 15;
    const int m0   = blockIdx.y * BM;
    const int n0   = blockIdx.x * BN;

    v8f acc[4][2];
#pragma unroll
    for (int i = 0; i < 4; ++i)
#pragma unroll
        for (int j = 0; j < 2; ++j) acc[i][j] = zero_v8f();

    // Prologue: stage K-slice 0 into buffer 0.
    stage_gemm_tile(X, W, gsm, gsm + BM * LDT, m0, n0, 0, tid);
    async_wait0();
    __syncthreads();

    for (int k0 = 0; k0 < D_MODELC; k0 += BK) {
        const int cur = (k0 / BK) & 1;
        _Float16* As = gsm + cur * GEMM_BUF_F16;
        _Float16* Ws = As + BM * LDT;

        // Kick off async DMA for the next K-slice into the other buffer.
        if (k0 + BK < D_MODELC) {
            _Float16* An = gsm + (cur ^ 1) * GEMM_BUF_F16;
            stage_gemm_tile(X, W, An, An + BM * LDT, m0, n0, k0 + BK, tid);
        }
        // Warm L2 two slices ahead of the async engine.
        if (k0 + 2 * BK < D_MODELC) {
            __builtin_prefetch(X + (size_t)(m0 + (tid >> 1)) * D_MODELC + k0 + 2 * BK, 0, 1);
            __builtin_prefetch(W + (size_t)(n0 + (tid >> 1)) * D_MODELC + k0 + 2 * BK, 0, 1);
        }

        // A fragments: row = lane%16; a[i]: K = hf*8+i (+16 for upper 8)
        v16h afrag[4];
#pragma unroll
        for (int sm = 0; sm < 4; ++sm) {
            const _Float16* r = As + (wm * 64 + sm * 16 + cl) * LDT;
            afrag[sm] = ld16(r + hf * 8, r + 16 + hf * 8);
        }
        // B fragments from W rows (W^T columns): b[i] = W[n][hf*8+i] (+16 for upper 8)
        v16h bfrag[2];
#pragma unroll
        for (int sn = 0; sn < 2; ++sn) {
            const _Float16* r = Ws + (wn * 32 + sn * 16 + cl) * LDT;
            bfrag[sn] = ld16(r + hf * 8, r + 16 + hf * 8);
        }
#pragma unroll
        for (int sm = 0; sm < 4; ++sm)
#pragma unroll
            for (int sn = 0; sn < 2; ++sn)
                acc[sm][sn] = wmma_f16f32(afrag[sm], bfrag[sn], acc[sm][sn]);

        // Drain this wave's async DMAs, then rendezvous before buffer swap.
        async_wait0();
        __syncthreads();
    }

    // Epilogue. C layout: VGPR r -> (M = r + 8*hf, N = lane%16).
#pragma unroll
    for (int sn = 0; sn < 2; ++sn) {
        int ng = n0 + wn * 32 + sn * 16 + cl;
        float bv = bias[ng];
#pragma unroll
        for (int sm = 0; sm < 4; ++sm) {
#pragma unroll
            for (int r = 0; r < 8; ++r) {
                int mg = m0 + wm * 64 + sm * 16 + r + 8 * hf;
                float val = acc[sm][sn][r] + bv;
                if (mode == 3) {
                    out_f[(size_t)mg * D_MODELC + ng] = val;
                } else {
                    int bb = mg / SEQC, ss = mg % SEQC;
                    int hh = ng >> 6, dk = ng & 63;
                    size_t head = (size_t)bb * NUM_HEADSC + hh;
                    if (mode == 2)
                        out_h[(head * D_KC + dk) * SEQC + ss] = (_Float16)val;
                    else
                        out_h[(head * SEQC + ss) * D_KC + dk] = (_Float16)val;
                }
            }
        }
    }
}

// ---------------------------------------------------------------------------
// Flash attention: per (b,h), 128 q-rows per block (8 waves x 16 rows),
// stream 128-key tiles (double-buffered async DMA); online softmax; the
// S matrix never touches HBM.
// ---------------------------------------------------------------------------
#define BQ  128
#define BKV 128
#define KLD 72   // K tile row stride (f16)
#define VLD 136  // V^T tile row stride
#define PLD 136  // per-wave P tile row stride
#define KV_BUF_F16 (BKV * KLD + D_KC * VLD)
#define FLASH_SMEM_F16 (2 * KV_BUF_F16 + 8 * 16 * PLD)

__device__ __forceinline__ void stage_kv_tile(
    const _Float16* __restrict__ Kh, const _Float16* __restrict__ Vt,
    _Float16* Ks, _Float16* Vs, size_t headOff, int kv0, int tid) {
#pragma unroll
    for (int c = 0; c < 4; ++c) {
        int chunk = tid + c * 256;
        int kr = chunk >> 3, kc = (chunk & 7) * 8;
        cp16_async(Kh + headOff + (size_t)(kv0 + kr) * D_KC + kc, Ks + kr * KLD + kc);
        int vr = chunk >> 4, vc = (chunk & 15) * 8;
        cp16_async(Vt + headOff + (size_t)vr * SEQC + kv0 + vc, Vs + vr * VLD + vc);
    }
}

__global__ __launch_bounds__(256) void flash_attn(
    const _Float16* __restrict__ Qh,   // [B,H,S,DK]
    const _Float16* __restrict__ Kh,   // [B,H,S,DK]
    const _Float16* __restrict__ Vt,   // [B,H,DK,S]
    const int* __restrict__ mask,      // [S,S]
    _Float16* __restrict__ ctx)        // [B,S,H*DK] f16
{
    extern __shared__ _Float16 smem[];
    _Float16* Ps = smem + 2 * KV_BUF_F16; // 8 x [16][PLD] (wave-private)

    const int tid  = threadIdx.x;
    const int lane = tid & 31;
    const int wv   = tid >> 5;
    const int hf   = lane >> 4;
    const int cl   = lane & 15;
    const int bh   = blockIdx.x;
    const int b    = bh / NUM_HEADSC;
    const int h    = bh % NUM_HEADSC;
    const int q0   = blockIdx.y * BQ;
    const size_t headOff = (size_t)bh * SEQC * D_KC; // same for Qh/Kh/Vt

    // Load Q A-fragments once (rows q0 + wv*16 + lane%16, 64 dk = 2 k-steps).
    const _Float16* qrow = Qh + headOff + (size_t)(q0 + wv * 16 + cl) * D_KC;
    v16h qf[2];
    qf[0] = ld16(qrow + hf * 8,      qrow + 16 + hf * 8);
    qf[1] = ld16(qrow + 32 + hf * 8, qrow + 48 + hf * 8);

    v8f oacc[4];
#pragma unroll
    for (int d = 0; d < 4; ++d) oacc[d] = zero_v8f();
    float mrow[8], lrow[8];
#pragma unroll
    for (int r = 0; r < 8; ++r) { mrow[r] = -3.0e38f; lrow[r] = 0.f; }

    _Float16* Pw = Ps + wv * 16 * PLD;
    const float scale = 0.125f; // 1/sqrt(64)

    // Prologue: stage KV tile 0 into buffer 0.
    stage_kv_tile(Kh, Vt, smem, smem + BKV * KLD, headOff, 0, tid);
    async_wait0();
    __syncthreads();

    for (int kv0 = 0; kv0 < SEQC; kv0 += BKV) {
        const int cur = (kv0 / BKV) & 1;
        _Float16* Ks = smem + cur * KV_BUF_F16;
        _Float16* Vs = Ks + BKV * KLD;

        // Async-stage the next KV tile into the other buffer.
        if (kv0 + BKV < SEQC) {
            _Float16* Kn = smem + (cur ^ 1) * KV_BUF_F16;
            stage_kv_tile(Kh, Vt, Kn, Kn + BKV * KLD, headOff, kv0 + BKV, tid);
        }

        // S = Q K^T over 8 key sub-tiles (2 WMMA k-steps each, dk=64).
        v8f sacc[8];
#pragma unroll
        for (int n = 0; n < 8; ++n) {
            v8f c = zero_v8f();
#pragma unroll
            for (int ks = 0; ks < 2; ++ks) {
                const _Float16* r = Ks + (n * 16 + cl) * KLD + ks * 32;
                v16h kb = ld16(r + hf * 8, r + 16 + hf * 8);
                c = wmma_f16f32(qf[ks], kb, c);
            }
            sacc[n] = c;
        }

        // Scale + mask + per-lane row max.
        float vmax[8];
#pragma unroll
        for (int r = 0; r < 8; ++r) vmax[r] = -3.0e38f;
#pragma unroll
        for (int n = 0; n < 8; ++n) {
#pragma unroll
            for (int r = 0; r < 8; ++r) {
                int qg = q0 + wv * 16 + r + 8 * hf;
                int kg = kv0 + n * 16 + cl;
                float s = sacc[n][r] * scale;
                if (mask[(size_t)qg * SEQC + kg] == 0) s = -1.0e9f;
                sacc[n][r] = s;
                vmax[r] = fmaxf(vmax[r], s);
            }
        }
        // Row reduce across the 16 lanes of each half (wave32: masks<16 stay in-half).
#pragma unroll
        for (int r = 0; r < 8; ++r) {
            float v = vmax[r];
            v = fmaxf(v, __shfl_xor(v, 1, 32));
            v = fmaxf(v, __shfl_xor(v, 2, 32));
            v = fmaxf(v, __shfl_xor(v, 4, 32));
            v = fmaxf(v, __shfl_xor(v, 8, 32));
            vmax[r] = v;
        }
        float alpha[8];
#pragma unroll
        for (int r = 0; r < 8; ++r) {
            float mn = fmaxf(mrow[r], vmax[r]);
            alpha[r] = __expf(mrow[r] - mn);
            mrow[r]  = mn;
        }
        // P = exp(S - m); stash into wave-private LDS for C->A relayout.
        float psum[8];
#pragma unroll
        for (int r = 0; r < 8; ++r) psum[r] = 0.f;
#pragma unroll
        for (int n = 0; n < 8; ++n) {
#pragma unroll
            for (int r = 0; r < 8; ++r) {
                float p = __expf(sacc[n][r] - mrow[r]);
                psum[r] += p;
                Pw[(r + 8 * hf) * PLD + n * 16 + cl] = (_Float16)p;
            }
        }
#pragma unroll
        for (int r = 0; r < 8; ++r) {
            float v = psum[r];
            v += __shfl_xor(v, 1, 32);
            v += __shfl_xor(v, 2, 32);
            v += __shfl_xor(v, 4, 32);
            v += __shfl_xor(v, 8, 32);
            lrow[r] = lrow[r] * alpha[r] + v;
        }
#pragma unroll
        for (int d = 0; d < 4; ++d)
#pragma unroll
            for (int r = 0; r < 8; ++r) oacc[d][r] *= alpha[r];

        // O += P V  (K-dim = 128 keys = 4 WMMA k-steps; accumulate via C operand).
#pragma unroll
        for (int ks = 0; ks < 4; ++ks) {
            const _Float16* pr = Pw + cl * PLD + ks * 32;
            v16h pa = ld16(pr + hf * 8, pr + 16 + hf * 8);
#pragma unroll
            for (int d = 0; d < 4; ++d) {
                const _Float16* vr = Vs + (d * 16 + cl) * VLD + ks * 32;
                v16h vb = ld16(vr + hf * 8, vr + 16 + hf * 8);
                oacc[d] = wmma_f16f32(pa, vb, oacc[d]);
            }
        }

        // Drain this wave's async DMAs, rendezvous, swap buffers.
        async_wait0();
        __syncthreads();
    }

    // Write context in concat-head layout [B,S,H*DK] f16.
#pragma unroll
    for (int d = 0; d < 4; ++d) {
#pragma unroll
        for (int r = 0; r < 8; ++r) {
            int sg = q0 + wv * 16 + r + 8 * hf;
            float v = oacc[d][r] / lrow[r];
            ctx[((size_t)b * SEQC + sg) * D_MODELC + h * D_KC + d * 16 + cl] =
                (_Float16)v;
        }
    }
}

// ---------------------------------------------------------------------------
// Host launcher
// ---------------------------------------------------------------------------
extern "C" void kernel_launch(void* const* d_in, const int* in_sizes, int n_in,
                              void* d_out, int out_size, void* d_ws, size_t ws_size,
                              hipStream_t stream) {
    const float* q    = (const float*)d_in[0];
    const float* k    = (const float*)d_in[1];
    const float* v    = (const float*)d_in[2];
    const int*   mask = (const int*)  d_in[3];
    const float* Wq   = (const float*)d_in[4];
    const float* bq   = (const float*)d_in[5];
    const float* Wk   = (const float*)d_in[6];
    const float* bk   = (const float*)d_in[7];
    const float* Wv   = (const float*)d_in[8];
    const float* bv   = (const float*)d_in[9];
    const float* Wo   = (const float*)d_in[10];
    const float* bo   = (const float*)d_in[11];
    float* out = (float*)d_out;

    const size_t NQ = (size_t)BATCHC * SEQC * D_MODELC; // 8,388,608
    const size_t NW = (size_t)D_MODELC * D_MODELC;      // 1,048,576

    _Float16* ws   = (_Float16*)d_ws;
    _Float16* q16  = ws; ws += NQ;
    _Float16* k16  = ws; ws += NQ;
    _Float16* v16  = ws; ws += NQ;
    _Float16* wq16 = ws; ws += NW;
    _Float16* wk16 = ws; ws += NW;
    _Float16* wv16 = ws; ws += NW;
    _Float16* wo16 = ws; ws += NW;
    _Float16* Qh   = ws; ws += NQ;
    _Float16* Khp  = ws; ws += NQ;
    _Float16* Vtp  = ws; ws += NQ;
    _Float16* ctx  = ws; ws += NQ;

    // fp32 -> f16
    cvt_f32_f16<<<(int)(NQ / 4 / 256), 256, 0, stream>>>(q, q16, (int)NQ);
    cvt_f32_f16<<<(int)(NQ / 4 / 256), 256, 0, stream>>>(k, k16, (int)NQ);
    cvt_f32_f16<<<(int)(NQ / 4 / 256), 256, 0, stream>>>(v, v16, (int)NQ);
    cvt_f32_f16<<<(int)(NW / 4 / 256), 256, 0, stream>>>(Wq, wq16, (int)NW);
    cvt_f32_f16<<<(int)(NW / 4 / 256), 256, 0, stream>>>(Wk, wk16, (int)NW);
    cvt_f32_f16<<<(int)(NW / 4 / 256), 256, 0, stream>>>(Wv, wv16, (int)NW);
    cvt_f32_f16<<<(int)(NW / 4 / 256), 256, 0, stream>>>(Wo, wo16, (int)NW);

    const size_t gemm_smem = (size_t)GEMM_SMEM_F16 * sizeof(_Float16);   // 40 KB
    const size_t flash_smem = (size_t)FLASH_SMEM_F16 * sizeof(_Float16); // ~104 KB

    // Q/K/V projections, scattering into head layouts (V transposed).
    dim3 gg(D_MODELC / BN, M_TOTAL / BM); // (8, 64)
    gemm_xwT<<<gg, 256, gemm_smem, stream>>>(q16, wq16, bq, Qh,  nullptr, 0);
    gemm_xwT<<<gg, 256, gemm_smem, stream>>>(k16, wk16, bk, Khp, nullptr, 1);
    gemm_xwT<<<gg, 256, gemm_smem, stream>>>(v16, wv16, bv, Vtp, nullptr, 2);

    // Flash attention.
    dim3 fg(BATCHC * NUM_HEADSC, SEQC / BQ); // (64, 16)
    flash_attn<<<fg, 256, flash_smem, stream>>>(Qh, Khp, Vtp, mask, ctx);

    // Output projection (f32 + bias) -> d_out.
    gemm_xwT<<<gg, 256, gemm_smem, stream>>>(ctx, wo16, bo, nullptr, out, 3);
}
